// Model_19731079758125
// MI455X (gfx1250) — matrix operand order repaired and checked by reading, not compiled, not run on hardware
//
#include <hip/hip_runtime.h>

// ---------------------------------------------------------------------------
// Fused NeRF MLP for gfx1250 (MI455X), f16 WMMA + async global->LDS weights.
//
//  - prepack kernel: f32 weights -> f16 in WMMA B-fragment layout (d_ws)
//  - fused kernel: 64 points/WG, activations ping-pong in LDS, weights
//    streamed per-K-slice via GLOBAL_LOAD_ASYNC_TO_LDS_B128 into a
//    double-buffered 2x16KB LDS region (copy overlaps WMMA compute),
//    v_wmma_f32_16x16x32_f16 for all matmuls, skip-concat free via layout.
//  - kt-loop unrolled by 2 so each slice uses a compile-time LDS buffer and
//    all fragment loads batch ahead of the 8-WMMA burst (partial dscnt waits).
//  - LDS = 82KB acts + 32KB weight buffers = 114KB -> 2 WGs per WGP.
// ---------------------------------------------------------------------------

typedef __attribute__((ext_vector_type(16))) _Float16 v16h;
typedef __attribute__((ext_vector_type(8)))  _Float16 v8h;
typedef __attribute__((ext_vector_type(8)))  float    v8f;

#define TM     64            // points per workgroup
#define PITCH  328           // halfs per activation row (320 used + 8 pad)
#define SLICE_HALFS 8192     // one k-slice = 16 fragments * 512 halfs (16KB)
#define SLICE_BYTES 16384
#define WOFF   (2 * TM * PITCH)   // half-offset of weight buffers in smem

// Packed-weight layout (halfs), per layer: KT*NT fragments of 512 halfs.
// Fragment(kt,nt): lane L, elem i -> W[kt*32 + (L<16?0:8) + (i&7) + (i<8?0:16)][nt*16 + (L&15)]
// KT = {2,8,8,8,8,10,8,8,8}; NT = {16 x8, 1}
#define PACKED_TOTAL 495616

struct WPtrs { const float* w[9]; };
struct BPtrs { const float* b[9]; };

__device__ __forceinline__ v8f wmma16x16x32(v16h a, v16h b, v8f c) {
  return __builtin_amdgcn_wmma_f32_16x16x32_f16(
      /*neg_a=*/false, a, /*neg_b=*/false, b,
      /*c_mod=*/(short)0, c, /*reuse_a=*/false, /*reuse_b=*/false);
}

// Async DMA global->LDS, tracked by ASYNCcnt. The instruction offset applies
// to BOTH the global and the LDS address (ISA 15.18 async pseudocode), so a
// 16KB cooperative copy is one address setup + 4 straight-line instructions.
__device__ __forceinline__ void async_copy16k(const _Float16* gsrc,
                                              unsigned int lds_byte, int tid) {
  const char* g = (const char*)gsrc + tid * 16;
  unsigned int d = lds_byte + tid * 16;
  asm volatile("global_load_async_to_lds_b128 %0, %1, off"               :: "v"(d), "v"(g) : "memory");
  asm volatile("global_load_async_to_lds_b128 %0, %1, off offset:4096"  :: "v"(d), "v"(g) : "memory");
  asm volatile("global_load_async_to_lds_b128 %0, %1, off offset:8192"  :: "v"(d), "v"(g) : "memory");
  asm volatile("global_load_async_to_lds_b128 %0, %1, off offset:12288" :: "v"(d), "v"(g) : "memory");
}
__device__ __forceinline__ void async_copy8k(const _Float16* gsrc,
                                             unsigned int lds_byte, int tid) {
  const char* g = (const char*)gsrc + tid * 16;
  unsigned int d = lds_byte + tid * 16;
  asm volatile("global_load_async_to_lds_b128 %0, %1, off"              :: "v"(d), "v"(g) : "memory");
  asm volatile("global_load_async_to_lds_b128 %0, %1, off offset:4096" :: "v"(d), "v"(g) : "memory");
}
__device__ __forceinline__ void wait_async0() {
  asm volatile("s_wait_asynccnt 0x0" ::: "memory");
}

// One k-slice of the layer GEMM: load 2 A-fragments + 4 B-fragments (batched
// ds_load_b128s), then 8 WMMAs into the persistent accumulators.
__device__ __forceinline__ void compute_slice(
    v8f acc[2][4], const _Float16* __restrict__ src, const _Float16* __restrict__ wk,
    int k0base, int mBase, int ntBase, int mr, int g, int lane) {
  const int k0 = k0base + g * 8;
  union AV { v16h v; v8h h[2]; } a0, a1;
  a0.h[0] = *(const v8h*)(src + (mBase + mr) * PITCH + k0);
  a0.h[1] = *(const v8h*)(src + (mBase + mr) * PITCH + k0 + 16);
  a1.h[0] = *(const v8h*)(src + (mBase + 16 + mr) * PITCH + k0);
  a1.h[1] = *(const v8h*)(src + (mBase + 16 + mr) * PITCH + k0 + 16);
  v16h b0 = *(const v16h*)(wk + (ntBase + 0) * 512 + lane * 16);
  v16h b1 = *(const v16h*)(wk + (ntBase + 1) * 512 + lane * 16);
  v16h b2 = *(const v16h*)(wk + (ntBase + 2) * 512 + lane * 16);
  v16h b3 = *(const v16h*)(wk + (ntBase + 3) * 512 + lane * 16);
  acc[0][0] = wmma16x16x32(a0.v, b0, acc[0][0]);
  acc[1][0] = wmma16x16x32(a1.v, b0, acc[1][0]);
  acc[0][1] = wmma16x16x32(a0.v, b1, acc[0][1]);
  acc[1][1] = wmma16x16x32(a1.v, b1, acc[1][1]);
  acc[0][2] = wmma16x16x32(a0.v, b2, acc[0][2]);
  acc[1][2] = wmma16x16x32(a1.v, b2, acc[1][2]);
  acc[0][3] = wmma16x16x32(a0.v, b3, acc[0][3]);
  acc[1][3] = wmma16x16x32(a1.v, b3, acc[1][3]);
}

// ---------------------------------------------------------------------------
// Prepack: f32 weights -> f16 fragments (zero-padded) in d_ws.
// ---------------------------------------------------------------------------
__global__ __launch_bounds__(256) void prepack(WPtrs wp, _Float16* __restrict__ packed) {
  constexpr int pko[10] = {0, 16384, 81920, 147456, 212992, 278528,
                           360448, 425984, 491520, 495616};
  constexpr int Ks[9]  = {63, 256, 256, 256, 256, 319, 256, 256, 256};
  constexpr int Ns[9]  = {256, 256, 256, 256, 256, 256, 256, 256, 4};
  constexpr int NTs[9] = {16, 16, 16, 16, 16, 16, 16, 16, 1};

  int idx = blockIdx.x * 256 + threadIdx.x;
  if (idx >= PACKED_TOTAL) return;

  int l = 0;
  while (idx >= pko[l + 1]) ++l;
  int rel  = idx - pko[l];
  int frag = rel >> 9;
  int r    = rel & 511;
  int lane = r >> 4;
  int i    = r & 15;
  int kt   = frag / NTs[l];
  int nt   = frag - kt * NTs[l];
  int k    = kt * 32 + ((lane >= 16) ? 8 : 0) + (i & 7) + ((i >= 8) ? 16 : 0);
  int n    = nt * 16 + (lane & 15);

  float v = 0.0f;
  if (k < Ks[l] && n < Ns[l]) v = wp.w[l][k * Ns[l] + n];
  packed[idx] = (_Float16)v;
}

// ---------------------------------------------------------------------------
// Fused MLP. Block = 256 threads = 8 waves. Grid = npts / 64.
// LDS: act0[64][328] | act1[64][328] | wbuf0[8192] | wbuf1[8192]  (f16)
// ---------------------------------------------------------------------------
__global__ __launch_bounds__(256) void nerf_fused(
    const float* __restrict__ point, float* __restrict__ out,
    const _Float16* __restrict__ packed, BPtrs bp, int npts) {
  constexpr int KTs[8] = {2, 8, 8, 8, 8, 10, 8, 8};
  constexpr int pko[9] = {0, 16384, 81920, 147456, 212992, 278528,
                          360448, 425984, 491520};

  extern __shared__ _Float16 smem[];
  _Float16* act0 = smem;
  _Float16* act1 = smem + TM * PITCH;
  _Float16* wbuf0 = smem + WOFF;
  _Float16* wbuf1 = smem + WOFF + SLICE_HALFS;
  // LDS byte offset of weight buffers (low 32 bits of generic LDS pointer)
  const unsigned int ldsW = (unsigned int)(unsigned long long)(smem + WOFF);

  const int tid    = threadIdx.x;
  const int lane   = tid & 31;
  const int wv     = tid >> 5;
  const int g      = lane >> 4;   // lane group (0: lanes 0-15, 1: lanes 16-31)
  const int mr     = lane & 15;
  const int ptBase = blockIdx.x * TM;

  // ---- prefetch layer 0, slice 0 while we compute the embedding ----
  async_copy16k(packed, ldsW, tid);

  // ---- positional encoding: emb[m][j] -> cols 256..318 of BOTH buffers ----
  for (int idx = tid; idx < TM * 63; idx += 256) {
    int m  = idx / 63;
    int j  = idx - m * 63;
    int c  = j / 21;
    int jj = j - c * 21;
    int pt = ptBase + m;
    float p = (pt < npts) ? point[pt * 3 + c] : 0.0f;
    float v;
    if (jj == 0)       v = p;
    else if (jj <= 10) v = sinf(p * exp2f((float)(jj - 1)));
    else               v = cosf(p * exp2f((float)(jj - 11)));
    _Float16 h = (_Float16)v;
    act0[m * PITCH + 256 + j] = h;
    act1[m * PITCH + 256 + j] = h;
  }
  if (tid < TM) {   // zero pad column 319 (K-padding for layers 0 and 5)
    act0[tid * PITCH + 319] = (_Float16)0.0f;
    act1[tid * PITCH + 319] = (_Float16)0.0f;
  }

  // Wave tiling: 2 M-tiles x 4 N-tiles per wave (32x64 output block).
  const int mBase  = (wv & 1) * 32;
  const int ntBase = (wv >> 1) * 4;

  // ---- hidden layers 0..7 (N = 256), k-slice double-buffered pipeline ----
  // All KT values are even, so unroll by 2: even slice -> wbuf0, odd -> wbuf1.
  for (int l = 0; l < 8; ++l) {
    const _Float16* src = (l & 1) ? act1 : act0;
    _Float16*       dst = (l & 1) ? act0 : act1;
    const int colOff = (l == 0) ? 256 : 0;   // layer 0 reads the embedding
    const int ktN    = KTs[l];
    const _Float16* layerW = packed + pko[l];

    v8f acc[2][4] = {};
    for (int kt = 0; kt < ktN; kt += 2) {
      // --- even slice: compute from wbuf0, stream slice kt+1 -> wbuf1 ---
      wait_async0();        // my copies for slice kt landed in LDS
      __syncthreads();      // everyone's copies landed; everyone done reading
                            // slice kt-1 and (at kt==0) prev-layer acts
      async_copy16k(layerW + (kt + 1) * SLICE_HALFS, ldsW + SLICE_BYTES, tid);
      compute_slice(acc, src, wbuf0, colOff + kt * 32, mBase, ntBase, mr, g, lane);

      // --- odd slice: compute from wbuf1, stream kt+2 / next layer -> wbuf0
      wait_async0();
      __syncthreads();
      if (kt + 2 < ktN)
        async_copy16k(layerW + (kt + 2) * SLICE_HALFS, ldsW, tid);
      else if (l < 7)
        async_copy16k(packed + pko[l + 1], ldsW, tid);
      else
        async_copy8k(packed + 491520, ldsW, tid);     // L8 weights (8KB)
      compute_slice(acc, src, wbuf1, colOff + (kt + 1) * 32, mBase, ntBase, mr, g, lane);
    }

    // epilogue: bias + ReLU, f32 -> f16, store to dst (cols 0..255)
    const float* bias = bp.b[l];
#pragma unroll
    for (int mt = 0; mt < 2; ++mt) {
#pragma unroll
      for (int t = 0; t < 4; ++t) {
        const int col = (ntBase + t) * 16 + mr;
        const float bv = bias[col];
#pragma unroll
        for (int r = 0; r < 8; ++r) {
          const int row = mBase + mt * 16 + g * 8 + r;
          float v = acc[mt][t][r] + bv;
          v = v > 0.0f ? v : 0.0f;
          dst[row * PITCH + col] = (_Float16)v;
        }
      }
    }
  }

  // ---- output layer 8: K=256 (src = act0), N=4 (padded to 16) ----
  wait_async0();       // L8 weights (8KB, prefetched during L7) are in wbuf0
  __syncthreads();     // L7 epilogue stores drained for all waves

  if (wv < 4) {        // 4 M-tiles, wave-uniform branch (EXEC all ones)
    const _Float16* srcf = act0;
    const int mB = wv * 16;
    v8f acc = {};
#pragma unroll
    for (int kt = 0; kt < 8; ++kt) {
      const int k0 = kt * 32 + g * 8;
      union AV { v16h v; v8h h[2]; } a;
      a.h[0] = *(const v8h*)(srcf + (mB + mr) * PITCH + k0);
      a.h[1] = *(const v8h*)(srcf + (mB + mr) * PITCH + k0 + 16);
      v16h b = *(const v16h*)(wbuf0 + kt * 512 + lane * 16);
      acc = wmma16x16x32(a.v, b, acc);
    }
    if (mr < 4) {      // only cols 0..3 are real outputs
      const float bv = bp.b[8][mr];
#pragma unroll
      for (int r = 0; r < 8; ++r) {
        int pt = ptBase + mB + g * 8 + r;
        if (pt < npts) out[pt * 4 + mr] = acc[r] + bv;
      }
    }
  }
}

// ---------------------------------------------------------------------------
extern "C" void kernel_launch(void* const* d_in, const int* in_sizes, int n_in,
                              void* d_out, int out_size, void* d_ws, size_t ws_size,
                              hipStream_t stream) {
  const float* point = (const float*)d_in[0];
  WPtrs wp;
  BPtrs bp;
  for (int i = 0; i < 9; ++i) {
    wp.w[i] = (const float*)d_in[1 + 2 * i];
    bp.b[i] = (const float*)d_in[2 + 2 * i];
  }
  _Float16* packed = (_Float16*)d_ws;   // needs PACKED_TOTAL*2 ~ 0.95 MB of ws
  const int npts = in_sizes[0] / 3;

  // 1) pack weights to f16 fragment layout (L2-resident afterwards)
  prepack<<<(PACKED_TOTAL + 255) / 256, 256, 0, stream>>>(wp, packed);

  // 2) fused MLP: 114KB LDS -> 2 workgroups per WGP
  const size_t shmem = (size_t)(WOFF + 2 * SLICE_HALFS) * sizeof(_Float16);
  (void)hipFuncSetAttribute((const void*)nerf_fused,
                            hipFuncAttributeMaxDynamicSharedMemorySize,
                            (int)shmem);
  const int nblk = (npts + TM - 1) / TM;
  nerf_fused<<<nblk, 256, shmem, stream>>>(point, (float*)d_out, packed, bp, npts);
}
